// ArcPredictionModel_52578989638037
// MI455X (gfx1250) — matrix operand order repaired and checked by reading, not compiled, not run on hardware
//
#include <hip/hip_runtime.h>
#include <hip/hip_bf16.h>

// ---------------------------------------------------------------------------
// ArcPredictionModel for MI455X (gfx1250, wave32, WMMA)
//
// Pipeline:
//   1. convert weights f32 -> f16 (Wih, W1, W2, Wb)
//   2. embed -> z16 [T,128] (f16)
//   3. per layer: Gi = z16 @ Wih16^T + bih   (WMMA f16->f32)    [2 dirs]
//                 sequential GRU scan (f32 VALU, 2 persistent blocks)
//   4. A16 = relu(z W1^T + b1), B16 = relu(z W2^T + b2)          (WMMA)
//      C16 = A16 @ Wb16                                          (WMMA)
//   5. score[n] = dot(C16[src], B16[dst]) + bb   (1M arcs, L2 gathers,
//      v_fma_mix_f32 inner product, dual accumulators)
// ---------------------------------------------------------------------------

typedef __attribute__((ext_vector_type(16))) _Float16 v16h;
typedef __attribute__((ext_vector_type(8)))  _Float16 v8h;
typedef __attribute__((ext_vector_type(8)))  float    v8f;
typedef __attribute__((ext_vector_type(4)))  float    v4f;

#define T_SEQ   2048
#define HID     128
#define HD      64
#define G3      192     // 3*HD gates
#define WAVE    32

// ---------------- WMMA fragment loaders (contiguous k order, same in A and B) ----------------

// A-matrix 16x32 f16 tile from row-major src (leading dim lda).
// lane L: m = m0 + (L&15); k = k0 + (L>>4)*16 + e
// -> two contiguous 8-halfword (16B) loads per lane.
__device__ __forceinline__ v16h load_a_tile(const _Float16* src, int lda, int m0, int k0) {
    int lane  = threadIdx.x & (WAVE - 1);
    int m     = m0 + (lane & 15);
    int kbase = k0 + ((lane >> 4) << 4);
    const _Float16* row = src + (size_t)m * lda;
    v8h lo = *(const v8h*)(row + kbase);
    v8h hi = *(const v8h*)(row + kbase + 8);
    v16h a;
#pragma unroll
    for (int e = 0; e < 8; ++e) { a[e] = lo[e]; a[8 + e] = hi[e]; }
    return a;
}

// B-matrix 32x16 tile where B[k][n] = W[n][k] (computes Z @ W^T); W row-major [N][K].
// lane L: n = n0 + (L&15); k = k0 + (L>>4)*16 + e -> two contiguous 16B loads.
__device__ __forceinline__ v16h load_bT_tile(const _Float16* W, int ldw, int n0, int k0) {
    int lane  = threadIdx.x & (WAVE - 1);
    int n     = n0 + (lane & 15);
    int kbase = k0 + ((lane >> 4) << 4);
    const _Float16* row = W + (size_t)n * ldw;
    v8h lo = *(const v8h*)(row + kbase);
    v8h hi = *(const v8h*)(row + kbase + 8);
    v16h b;
#pragma unroll
    for (int e = 0; e < 8; ++e) { b[e] = lo[e]; b[8 + e] = hi[e]; }
    return b;
}

// B-matrix 32x16 tile, non-transposed: B[k][n] = W[k][n]; W row-major [K][N].
__device__ __forceinline__ v16h load_b_tile(const _Float16* W, int ldw, int n0, int k0) {
    int lane  = threadIdx.x & (WAVE - 1);
    int n     = n0 + (lane & 15);
    int kbase = k0 + ((lane >> 4) << 4);
    v16h b;
#pragma unroll
    for (int e = 0; e < 16; ++e) b[e] = W[(size_t)(kbase + e) * ldw + n];
    return b;
}

// ---------------- simple prep kernels ----------------

__global__ void f32_to_f16_kernel(const float* __restrict__ src,
                                  _Float16* __restrict__ dst, int n) {
    int i = blockIdx.x * blockDim.x + threadIdx.x;
    if (i < n) dst[i] = (_Float16)src[i];
}

// one block per note (128 threads): z16[t] = concat(E_pitch[p], E_dur[d], E_met[m])
__global__ void embed_kernel(const int* __restrict__ nf,
                             const float* __restrict__ Ep,
                             const float* __restrict__ Ed,
                             const float* __restrict__ Em,
                             _Float16* __restrict__ Z16) {
    int t = blockIdx.x;
    int j = threadIdx.x;                 // 0..127
    int p = nf[t * 4 + 0];
    int d = nf[t * 4 + 2];
    int m = nf[t * 4 + 3];
    float v;
    if (j < 96)        v = Ep[p * 96 + j];
    else if (j < 120)  v = Ed[d * 24 + (j - 96)];
    else               v = Em[m * 8  + (j - 120)];
    Z16[(size_t)t * HID + j] = (_Float16)v;
}

// ---------------- GEMM kernels (WMMA f16 -> f32 accum) ----------------

// Gi[dir][t][0:192] = Z16[t] @ Wih16[dir]^T + bih[dir]
// 8 waves/block, one 16x16 output tile per wave. tiles = (T/16)*12*2.
__global__ void gi_gemm_kernel(const _Float16* __restrict__ Z16,
                               const _Float16* __restrict__ Wih16,  // [2][192][128]
                               const float* __restrict__ bih,       // [2][192]
                               float* __restrict__ Gi, int Tn) {
    int mtiles = Tn / 16;
    int total  = mtiles * (G3 / 16) * 2;
    int tile   = blockIdx.x * (blockDim.x / WAVE) + (threadIdx.x / WAVE);
    if (tile >= total) return;
    int dir = tile / (mtiles * (G3 / 16));
    int rem = tile % (mtiles * (G3 / 16));
    int m0  = (rem / (G3 / 16)) * 16;
    int n0  = (rem % (G3 / 16)) * 16;
    const _Float16* W = Wih16 + (size_t)dir * G3 * HID;

    v8f c = {};
#pragma unroll
    for (int k0 = 0; k0 < HID; k0 += 32) {
        v16h a = load_a_tile(Z16, HID, m0, k0);
        v16h b = load_bT_tile(W, HID, n0, k0);
        c = __builtin_amdgcn_wmma_f32_16x16x32_f16(false, a, false, b,
                                                   (short)0, c, false, false);
    }
    int lane  = threadIdx.x & (WAVE - 1);
    int n     = n0 + (lane & 15);
    int mbase = m0 + ((lane >> 4) << 3);
    float bias = bih[dir * G3 + n];
    float* D = Gi + (size_t)dir * Tn * G3;
#pragma unroll
    for (int r = 0; r < 8; ++r) D[(size_t)(mbase + r) * G3 + n] = c[r] + bias;
}

// Out16 = relu(Z16 @ W16^T + bias) (head projections), 128x128 weights.
__global__ void head_gemm_kernel(const _Float16* __restrict__ Z16,
                                 const _Float16* __restrict__ W16,
                                 const float* __restrict__ bias,
                                 _Float16* __restrict__ Out16, int Tn) {
    int total = (Tn / 16) * (HID / 16);
    int tile  = blockIdx.x * (blockDim.x / WAVE) + (threadIdx.x / WAVE);
    if (tile >= total) return;
    int m0 = (tile / (HID / 16)) * 16;
    int n0 = (tile % (HID / 16)) * 16;

    v8f c = {};
#pragma unroll
    for (int k0 = 0; k0 < HID; k0 += 32) {
        v16h a = load_a_tile(Z16, HID, m0, k0);
        v16h b = load_bT_tile(W16, HID, n0, k0);
        c = __builtin_amdgcn_wmma_f32_16x16x32_f16(false, a, false, b,
                                                   (short)0, c, false, false);
    }
    int lane  = threadIdx.x & (WAVE - 1);
    int n     = n0 + (lane & 15);
    int mbase = m0 + ((lane >> 4) << 3);
    float bv  = bias[n];
#pragma unroll
    for (int r = 0; r < 8; ++r)
        Out16[(size_t)(mbase + r) * HID + n] = (_Float16)fmaxf(c[r] + bv, 0.0f);
}

// C16 = A16 @ Wb16 (Wb not transposed: C[t][k] = sum_h A[t][h]*Wb[h][k])
__global__ void bilin_gemm_kernel(const _Float16* __restrict__ A16,
                                  const _Float16* __restrict__ Wb16,
                                  _Float16* __restrict__ C16, int Tn) {
    int total = (Tn / 16) * (HID / 16);
    int tile  = blockIdx.x * (blockDim.x / WAVE) + (threadIdx.x / WAVE);
    if (tile >= total) return;
    int m0 = (tile / (HID / 16)) * 16;
    int n0 = (tile % (HID / 16)) * 16;

    v8f c = {};
#pragma unroll
    for (int k0 = 0; k0 < HID; k0 += 32) {
        v16h a = load_a_tile(A16, HID, m0, k0);
        v16h b = load_b_tile(Wb16, HID, n0, k0);
        c = __builtin_amdgcn_wmma_f32_16x16x32_f16(false, a, false, b,
                                                   (short)0, c, false, false);
    }
    int lane  = threadIdx.x & (WAVE - 1);
    int n     = n0 + (lane & 15);
    int mbase = m0 + ((lane >> 4) << 3);
#pragma unroll
    for (int r = 0; r < 8; ++r)
        C16[(size_t)(mbase + r) * HID + n] = (_Float16)c[r];
}

// ---------------- sequential GRU scan (f32, latency-bound) ----------------

__device__ __forceinline__ float sigmoidf_(float x) {
    return 1.0f / (1.0f + __expf(-x));
}

// 2 blocks (dir 0 = forward, dir 1 = backward), 192 threads each.
// Thread j owns gate row j: keeps Whh row (64 f32) in registers.
__global__ void gru_scan_kernel(const float* __restrict__ Gi,    // [2][T][192] (includes bih)
                                const float* __restrict__ Whh,   // [2][192][64] this layer
                                const float* __restrict__ bhh,   // [2][192]    this layer
                                _Float16* __restrict__ Zout16,   // [T][128]
                                int Tn) {
    int dir = blockIdx.x;
    int j   = threadIdx.x;               // 0..191
    __shared__ float h[HD];
    __shared__ float g[G3];
    __shared__ float gi_s[G3];

    float w[HD];
    const float* wrow = Whh + (size_t)(dir * G3 + j) * HD;
#pragma unroll
    for (int k = 0; k < HD; k += 4) {
        v4f v = *(const v4f*)(wrow + k);
        w[k] = v[0]; w[k+1] = v[1]; w[k+2] = v[2]; w[k+3] = v[3];
    }
    float bj = bhh[dir * G3 + j];
    if (j < HD) h[j] = 0.0f;
    __syncthreads();

    const float* GiD = Gi + (size_t)dir * Tn * G3;
    for (int s = 0; s < Tn; ++s) {
        int t = dir ? (Tn - 1 - s) : s;
        gi_s[j] = GiD[(size_t)t * G3 + j];
        float acc = bj;
#pragma unroll
        for (int k = 0; k < HD; ++k) acc = fmaf(w[k], h[k], acc);
        g[j] = acc;
        __syncthreads();
        if (j < HD) {
            float r  = sigmoidf_(gi_s[j]        + g[j]);
            float zg = sigmoidf_(gi_s[HD + j]   + g[HD + j]);
            float nn = tanhf    (gi_s[2*HD + j] + r * g[2*HD + j]);
            float hn = (1.0f - zg) * nn + zg * h[j];
            h[j] = hn;
            Zout16[(size_t)t * HID + dir * HD + j] = (_Float16)hn;
        }
        __syncthreads();
    }
}

// ---------------- arc scoring: 1M gathered dot products ----------------
// f16 x f16 -> f32 mixed FMA (v_fma_mix_f32), two accumulators for ILP.

__global__ void score_kernel(const int* __restrict__ arcs,      // [N][2]
                             const _Float16* __restrict__ Cn,   // [T][128]
                             const _Float16* __restrict__ Bn,   // [T][128]
                             const float* __restrict__ bb,
                             float* __restrict__ out, int n) {
    int i = blockIdx.x * blockDim.x + threadIdx.x;
    if (i >= n) return;
    int src = arcs[2 * i + 0];
    int dst = arcs[2 * i + 1];
    const _Float16* c = Cn + (size_t)src * HID;
    const _Float16* b = Bn + (size_t)dst * HID;
    __builtin_prefetch(b, 0, 0);   // global_prefetch_b8 of dst row while src loads issue
    float acc0 = 0.0f, acc1 = 0.0f;
#pragma unroll
    for (int j = 0; j < HID / 8; ++j) {
        v8h cv = *(const v8h*)(c + 8 * j);
        v8h bv = *(const v8h*)(b + 8 * j);
#pragma unroll
        for (int e = 0; e < 4; ++e) {
            acc0 = fmaf((float)cv[2 * e],     (float)bv[2 * e],     acc0);
            acc1 = fmaf((float)cv[2 * e + 1], (float)bv[2 * e + 1], acc1);
        }
    }
    out[i] = acc0 + acc1 + bb[0];
}

// ---------------------------------------------------------------------------

extern "C" void kernel_launch(void* const* d_in, const int* in_sizes, int n_in,
                              void* d_out, int out_size, void* d_ws, size_t ws_size,
                              hipStream_t stream) {
    const int*   nf   = (const int*)  d_in[0];
    const int*   arcs = (const int*)  d_in[1];
    const float* Ep   = (const float*)d_in[2];
    const float* Ed   = (const float*)d_in[3];
    const float* Em   = (const float*)d_in[4];
    const float* Wih  = (const float*)d_in[5];   // [2][2][192][128]
    const float* Whh  = (const float*)d_in[6];   // [2][2][192][64]
    const float* bih  = (const float*)d_in[7];   // [2][2][192]
    const float* bhh  = (const float*)d_in[8];   // [2][2][192]
    const float* W1   = (const float*)d_in[9];
    const float* b1   = (const float*)d_in[10];
    const float* W2   = (const float*)d_in[11];
    const float* b2   = (const float*)d_in[12];
    const float* Wb   = (const float*)d_in[13];  // [1][128][128]
    const float* bb   = (const float*)d_in[14];  // [1]
    float* out = (float*)d_out;

    const int Tn    = in_sizes[0] / 4;           // 2048
    const int nArcs = in_sizes[1] / 2;           // 1,000,000

    // workspace carve-out (256B aligned slices)
    char*  ws  = (char*)d_ws;
    size_t off = 0;
    auto carve = [&](size_t bytes) -> void* {
        void* p = ws + off;
        off += (bytes + 255) & ~(size_t)255;
        return p;
    };
    _Float16* z16a   = (_Float16*)carve((size_t)Tn * HID * 2);
    _Float16* z16b   = (_Float16*)carve((size_t)Tn * HID * 2);
    _Float16* Wih16  = (_Float16*)carve((size_t)2 * 2 * G3 * HID * 2);
    _Float16* W116   = (_Float16*)carve((size_t)HID * HID * 2);
    _Float16* W216   = (_Float16*)carve((size_t)HID * HID * 2);
    _Float16* Wb16   = (_Float16*)carve((size_t)HID * HID * 2);
    float*    Gi     = (float*)   carve((size_t)2 * Tn * G3 * 4);
    _Float16* A16    = (_Float16*)carve((size_t)Tn * HID * 2);
    _Float16* B16    = (_Float16*)carve((size_t)Tn * HID * 2);
    _Float16* C16    = (_Float16*)carve((size_t)Tn * HID * 2);

    // 1) weight conversions
    {
        int n = 2 * 2 * G3 * HID;
        f32_to_f16_kernel<<<(n + 255) / 256, 256, 0, stream>>>(Wih, Wih16, n);
        n = HID * HID;
        f32_to_f16_kernel<<<(n + 255) / 256, 256, 0, stream>>>(W1, W116, n);
        f32_to_f16_kernel<<<(n + 255) / 256, 256, 0, stream>>>(W2, W216, n);
        f32_to_f16_kernel<<<(n + 255) / 256, 256, 0, stream>>>(Wb, Wb16, n);
    }

    // 2) embeddings -> z16a
    embed_kernel<<<Tn, HID, 0, stream>>>(nf, Ep, Ed, Em, z16a);

    // 3) GRU layers (ping-pong z16a <-> z16b)
    _Float16* zin  = z16a;
    _Float16* zout = z16b;
    for (int l = 0; l < 2; ++l) {
        int tiles  = (Tn / 16) * (G3 / 16) * 2;
        int blocks = (tiles + 7) / 8;
        gi_gemm_kernel<<<blocks, 256, 0, stream>>>(
            zin, Wih16 + (size_t)l * 2 * G3 * HID, bih + (size_t)l * 2 * G3, Gi, Tn);
        gru_scan_kernel<<<2, G3, 0, stream>>>(
            Gi, Whh + (size_t)l * 2 * G3 * HD, bhh + (size_t)l * 2 * G3, zout, Tn);
        _Float16* tmp = zin; zin = zout; zout = tmp;
    }
    _Float16* zfinal = zin;   // after 2 swaps -> z16a

    // 4) per-note tables
    {
        int tiles  = (Tn / 16) * (HID / 16);
        int blocks = (tiles + 7) / 8;
        head_gemm_kernel<<<blocks, 256, 0, stream>>>(zfinal, W116, b1, A16, Tn);
        head_gemm_kernel<<<blocks, 256, 0, stream>>>(zfinal, W216, b2, B16, Tn);
        bilin_gemm_kernel<<<blocks, 256, 0, stream>>>(A16, Wb16, C16, Tn);
    }

    // 5) arc scores
    score_kernel<<<(nArcs + 255) / 256, 256, 0, stream>>>(arcs, C16, B16, bb, out, nArcs);

    (void)n_in; (void)out_size; (void)ws_size;
}
